// LSTM_12317966205561
// MI455X (gfx1250) — matrix operand order repaired and checked by reading, not compiled, not run on hardware
//
#include <hip/hip_runtime.h>
#include <hip/hip_bf16.h>
#include <math.h>

// ---------------------------------------------------------------------------
// MI455X (gfx1250) 2-layer LSTM.
//  Phase 1: fp32 -> f16 conversion of activations/weights (streaming).
//  Phase 2: big WMMA GEMM xw = A @ Brow^T + b_ih + b_hh   (v_wmma_f32_16x16x32_f16)
//  Phase 3: persistent single-workgroup recurrence, w_hh resident in LDS
//           (288KB padded f16), 48 WMMAs/wave/step, gates fused in registers.
// ---------------------------------------------------------------------------

typedef __attribute__((ext_vector_type(16))) _Float16 v16h;
typedef __attribute__((ext_vector_type(8)))  _Float16 v8h;
typedef __attribute__((ext_vector_type(8)))  float    v8f;

union F16x16 { v16h v; v8h h[2]; };

#define WMMA_F16(a, b, c) \
    __builtin_amdgcn_wmma_f32_16x16x32_f16(false, (a), false, (b), (short)0, (c), false, false)

__device__ __forceinline__ float sigmoidf_fast(float x) {
    return 1.0f / (1.0f + __expf(-x));
}

// -------------------------- fp32 -> f16 convert ----------------------------
__global__ void k_f32_to_f16(const float* __restrict__ in, _Float16* __restrict__ out, int n) {
    int i = blockIdx.x * blockDim.x + threadIdx.x;
    if (i < n) out[i] = (_Float16)in[i];
}

// Convert [rows, kin] fp32 -> [rows, kout] f16 with zero padding in K.
__global__ void k_convert_pad(const float* __restrict__ in, _Float16* __restrict__ out,
                              int rows, int kin, int kout) {
    int i = blockIdx.x * blockDim.x + threadIdx.x;
    int n = rows * kout;
    if (i >= n) return;
    int r = i / kout, k = i - r * kout;
    out[i] = (k < kin) ? (_Float16)in[r * kin + k] : (_Float16)0.f;
}

// ------------------------------ WMMA GEMM ----------------------------------
// Cout[m][n] = sum_k A[m][k]*Brow[n][k] + bias0[n] + bias1[n]
// A: [M,K] f16 row-major, Brow: [N,K] f16 row-major (i.e. B = Brow^T).
// One wave computes a 16(M) x 48(N) tile (3 WMMA accumulators, A reuse x3).
// Requires M%16==0, N%48==0, K%32==0.
__global__ void wmma_gemm_bias(const _Float16* __restrict__ A,
                               const _Float16* __restrict__ Brow,
                               const float* __restrict__ bias0,
                               const float* __restrict__ bias1,
                               float* __restrict__ Cout,
                               int M, int N, int K) {
    const int lane = threadIdx.x & 31;
    const int wave = threadIdx.x >> 5;
    const int wid  = blockIdx.x * (blockDim.x >> 5) + wave;
    const int ngroups = N / 48;
    const int mt = wid / ngroups;
    const int ng = wid - mt * ngroups;
    if (mt * 16 >= M) return;

    const int l15  = lane & 15;
    const int hi   = lane >> 4;            // 0 for lanes 0-15, 1 for 16-31
    const int kb_a = hi * 8;               // A half-chunk base (ISA 16-bit A layout)
    const int kb_b = hi * 8;               // B half-chunk base

    const _Float16* arow = A + (size_t)(mt * 16 + l15) * K;
    const _Float16* b0   = Brow + (size_t)(ng * 48 + l15) * K;
    const _Float16* b1   = b0 + (size_t)16 * K;
    const _Float16* b2   = b0 + (size_t)32 * K;

    v8f acc0 = {}, acc1 = {}, acc2 = {};

    for (int k = 0; k < K; k += 32) {
        __builtin_prefetch(arow + k + 256, 0, 1);   // global_prefetch_b8
        __builtin_prefetch(b0 + k + 256, 0, 1);
        F16x16 a, f0, f1, f2;
        a.h[0]  = *(const v8h*)(arow + k + kb_a);
        a.h[1]  = *(const v8h*)(arow + k + kb_a + 16);
        f0.h[0] = *(const v8h*)(b0 + k + kb_b);
        f0.h[1] = *(const v8h*)(b0 + k + kb_b + 16);
        f1.h[0] = *(const v8h*)(b1 + k + kb_b);
        f1.h[1] = *(const v8h*)(b1 + k + kb_b + 16);
        f2.h[0] = *(const v8h*)(b2 + k + kb_b);
        f2.h[1] = *(const v8h*)(b2 + k + kb_b + 16);
        acc0 = WMMA_F16(a.v, f0.v, acc0);
        acc1 = WMMA_F16(a.v, f1.v, acc1);
        acc2 = WMMA_F16(a.v, f2.v, acc2);
    }

    const int ncol = ng * 48 + l15;
    const float bz0 = bias0[ncol]      + bias1[ncol];
    const float bz1 = bias0[ncol + 16] + bias1[ncol + 16];
    const float bz2 = bias0[ncol + 32] + bias1[ncol + 32];
    const int rbase = 8 * hi;
#pragma unroll
    for (int r = 0; r < 8; ++r) {
        size_t orow = (size_t)(mt * 16 + rbase + r) * N;
        Cout[orow + ncol]      = acc0[r] + bz0;
        Cout[orow + ncol + 16] = acc1[r] + bz1;
        Cout[orow + ncol + 32] = acc2[r] + bz2;
    }
}

// ------------------------- persistent LSTM recurrence ----------------------
// One workgroup, 12 waves (384 threads). Wave w owns hidden slice
// u in [16w, 16w+16). LDS: W2 = padded recurrent weights [4*192][192] f16
// (288KB), hb = h state [32][192] f16 (12KB). Per step each wave runs
// 6 K-slabs x 4 gates x 2 M-tiles = 48 v_wmma_f32_16x16x32_f16 and fuses
// the i/f/g/o nonlinearity + cell update fully in registers.
__global__ void lstm_recur(const float* __restrict__ xw,    // [(B*T),720] = x@Wih^T + b_ih + b_hh
                           const float* __restrict__ w_hh,  // [720,180] fp32
                           _Float16* __restrict__ y_f16,    // [B*T,192] padded (or null)
                           float* __restrict__ y_f32,       // [B*T,180] (or null)
                           float* __restrict__ h_out,       // [32*180]
                           float* __restrict__ c_out,       // [32*180]
                           int T) {
    extern __shared__ char smem[];
    _Float16* W2 = (_Float16*)smem;                       // [768][192]
    _Float16* hb = (_Float16*)(smem + 768 * 192 * 2);     // [32][192]

    const int tid  = threadIdx.x;
    const int lane = tid & 31;
    const int wave = tid >> 5;     // u-tile index 0..11

    // Build padded recurrent weight tile in LDS: row j = gate*192 + u.
    for (int idx = tid; idx < 768 * 192; idx += blockDim.x) {
        int j = idx / 192, k = idx - j * 192;
        int gate = j / 192, u = j - gate * 192;
        float v = 0.f;
        if (u < 180 && k < 180) v = w_hh[(gate * 180 + u) * 180 + k];
        W2[idx] = (_Float16)v;
    }
    for (int idx = tid; idx < 32 * 192; idx += blockDim.x) hb[idx] = (_Float16)0.f;
    __syncthreads();

    const int l15    = lane & 15;
    const int hi     = lane >> 4;
    const int kb_a   = hi * 8;
    const int kb_b   = hi * 8;
    const int rbase  = 8 * hi;
    const int u      = wave * 16 + l15;         // hidden unit owned by this lane column
    const bool uval  = (u < 180);

    v8f  cfr[2] = {v8f{}, v8f{}};               // cell state fragments (mtile 0/1)
    float hlast[2][8];
#pragma unroll
    for (int m = 0; m < 2; ++m)
#pragma unroll
        for (int r = 0; r < 8; ++r) hlast[m][r] = 0.f;

    for (int t = 0; t < T; ++t) {
        // C operand = precomputed input projection rows for timestep t.
        v8f acc[4][2];
#pragma unroll
        for (int g = 0; g < 4; ++g) {
            int gcol = g * 180 + u;
#pragma unroll
            for (int m = 0; m < 2; ++m) {
#pragma unroll
                for (int r = 0; r < 8; ++r) {
                    int b = m * 16 + rbase + r;
                    acc[g][m][r] = uval ? xw[((size_t)b * T + t) * 720 + gcol] : 0.f;
                }
            }
        }
        // gates += h @ W2^T  (K = 192, 6 slabs of 32)
#pragma unroll
        for (int ks = 0; ks < 6; ++ks) {
            F16x16 a0, a1;
            const _Float16* h0 = hb + l15 * 192 + ks * 32;
            const _Float16* h1 = hb + (l15 + 16) * 192 + ks * 32;
            a0.h[0] = *(const v8h*)(h0 + kb_a);
            a0.h[1] = *(const v8h*)(h0 + kb_a + 16);
            a1.h[0] = *(const v8h*)(h1 + kb_a);
            a1.h[1] = *(const v8h*)(h1 + kb_a + 16);
#pragma unroll
            for (int g = 0; g < 4; ++g) {
                F16x16 bf;
                const _Float16* wr = W2 + (size_t)(g * 192 + wave * 16 + l15) * 192 + ks * 32 + kb_b;
                bf.h[0] = *(const v8h*)(wr);
                bf.h[1] = *(const v8h*)(wr + 16);
                acc[g][0] = WMMA_F16(a0.v, bf.v, acc[g][0]);
                acc[g][1] = WMMA_F16(a1.v, bf.v, acc[g][1]);
            }
        }
        __syncthreads();   // all waves done reading hb for step t

        // i,f,g,o fusion + cell update, all in registers.
#pragma unroll
        for (int m = 0; m < 2; ++m) {
#pragma unroll
            for (int r = 0; r < 8; ++r) {
                float iv = sigmoidf_fast(acc[0][m][r]);
                float fv = sigmoidf_fast(acc[1][m][r]);
                float gv = tanhf(acc[2][m][r]);
                float ov = sigmoidf_fast(acc[3][m][r]);
                float cn = fv * cfr[m][r] + iv * gv;
                cfr[m][r] = cn;
                float hn = ov * tanhf(cn);
                hlast[m][r] = hn;
                int b = m * 16 + rbase + r;
                _Float16 hh = uval ? (_Float16)hn : (_Float16)0.f;
                hb[b * 192 + u] = hh;                               // pad stays zero
                if (y_f16) y_f16[((size_t)b * T + t) * 192 + u] = hh;
                if (y_f32 && uval) y_f32[((size_t)b * T + t) * 180 + u] = hn;
            }
        }
        __syncthreads();   // new h visible before next step's GEMM
    }

    if (uval) {
#pragma unroll
        for (int m = 0; m < 2; ++m)
#pragma unroll
            for (int r = 0; r < 8; ++r) {
                int b = m * 16 + rbase + r;
                h_out[b * 180 + u] = hlast[m][r];
                c_out[b * 180 + u] = cfr[m][r];
            }
    }
}

// ------------------------------- launcher ----------------------------------
extern "C" void kernel_launch(void* const* d_in, const int* in_sizes, int n_in,
                              void* d_out, int out_size, void* d_ws, size_t ws_size,
                              hipStream_t stream) {
    const float* x       = (const float*)d_in[0];   // [32,128,6400]
    const float* w_ih_l0 = (const float*)d_in[1];   // [720,6400]
    const float* w_hh_l0 = (const float*)d_in[2];   // [720,180]
    const float* b_ih_l0 = (const float*)d_in[3];
    const float* b_hh_l0 = (const float*)d_in[4];
    const float* w_ih_l1 = (const float*)d_in[5];   // [720,180]
    const float* w_hh_l1 = (const float*)d_in[6];   // [720,180]
    const float* b_ih_l1 = (const float*)d_in[7];
    const float* b_hh_l1 = (const float*)d_in[8];

    const int B = 32, T = 128, I = 6400, H = 180;
    const int M = B * T;            // 4096
    const int G = 4 * H;            // 720
    const int Kp = 192;             // H padded to WMMA K granularity

    char* p = (char*)d_ws;
    _Float16* x_f16    = (_Float16*)p; p += (size_t)M * I * 2;
    _Float16* wih0_f16 = (_Float16*)p; p += (size_t)G * I * 2;
    float*    xw       = (float*)p;    p += (size_t)M * G * 4;   // reused by both layers
    _Float16* y0_f16   = (_Float16*)p; p += (size_t)M * Kp * 2;
    _Float16* wih1_f16 = (_Float16*)p; p += (size_t)G * Kp * 2;

    float* y1 = (float*)d_out;                   // [32,128,180]
    float* hn = y1 + (size_t)B * T * H;          // [2,32,180]
    float* cn = hn + (size_t)2 * B * H;          // [2,32,180]

    // Phase 1: precision conversion.
    {
        int n = M * I;
        k_f32_to_f16<<<(n + 255) / 256, 256, 0, stream>>>(x, x_f16, n);
        n = G * I;
        k_f32_to_f16<<<(n + 255) / 256, 256, 0, stream>>>(w_ih_l0, wih0_f16, n);
        n = G * Kp;
        k_convert_pad<<<(n + 255) / 256, 256, 0, stream>>>(w_ih_l1, wih1_f16, G, H, Kp);
    }

    const int waves   = (M / 16) * (G / 48);     // 256 * 15 = 3840
    const int blocks  = waves / 4;               // 4 waves / 128-thread block
    const size_t lds  = (size_t)768 * 192 * 2 + (size_t)32 * 192 * 2;  // 307200 B

    // Layer 0: input projection (K=6400), then recurrence.
    wmma_gemm_bias<<<blocks, 128, 0, stream>>>(x_f16, wih0_f16, b_ih_l0, b_hh_l0,
                                               xw, M, G, I);
    lstm_recur<<<1, 384, lds, stream>>>(xw, w_hh_l0, y0_f16, nullptr,
                                        hn, cn, T);

    // Layer 1: input projection over padded y0 (K=192), then recurrence.
    wmma_gemm_bias<<<blocks, 128, 0, stream>>>(y0_f16, wih1_f16, b_ih_l1, b_hh_l1,
                                               xw, M, G, Kp);
    lstm_recur<<<1, 384, lds, stream>>>(xw, w_hh_l1, nullptr, y1,
                                        hn + (size_t)B * H, cn + (size_t)B * H, T);
}
